// MultiHeadLatentAttention_65060164600303
// MI455X (gfx1250) — compile-verified, matrix operand order checked by reading, not run on hardware
//
#include <hip/hip_runtime.h>
#include <hip/hip_bf16.h>

// ---------------------------------------------------------------------------
// MLA forward for MI455X (gfx1250, wave32, WMMA + double-buffered async LDS).
// ---------------------------------------------------------------------------

typedef __attribute__((ext_vector_type(16))) __bf16 v16bf;
typedef __attribute__((ext_vector_type(8)))  __bf16 v8bf;
typedef __attribute__((ext_vector_type(8)))  float  v8f;

#define B_     2
#define S_     2048
#define H_     2048
#define NH_    16
#define QLORA_ 1536
#define KVLORA_ 512
#define DNOPE_ 128
#define DROPE_ 64
#define DV_    128
#define DQK_   192
#define MTOT_  (B_ * S_)          // 4096
#define QK_SCALE 0.07216878364870322f   // 1/sqrt(192)

__device__ __forceinline__ __bf16 f2bf(float f) {
    union { float f; unsigned int u; } a; a.f = f;
    unsigned int u = a.u;
    u += 0x7fffu + ((u >> 16) & 1u);          // round-to-nearest-even
    union { unsigned short s; __bf16 b; } o;
    o.s = (unsigned short)(u >> 16);
    return o.b;
}

// bf16 fragment: elements 0..7 at p[0..7], 8..15 at p[16..23]
// (CDNA5 16-bit matrix VGPR layout; caller adds row*ld + kstep + kb).
__device__ __forceinline__ v16bf load_frag(const __bf16* p) {
    v8bf lo = *(const v8bf*)(p);
    v8bf hi = *(const v8bf*)(p + 16);
    return __builtin_shufflevector(lo, hi,
        0,1,2,3,4,5,6,7,8,9,10,11,12,13,14,15);
}

__device__ __forceinline__ v8f wmma_bf16(v16bf a, v16bf b, v8f c) {
    return __builtin_amdgcn_wmma_f32_16x16x32_bf16(
        false, a, false, b, (short)0, c, false, false);
}

// CDNA5 async tensor path: memory -> LDS without VGPR round trip (ASYNCcnt).
__device__ __forceinline__ void async_copy_b128(unsigned lds_off, const void* g) {
    asm volatile("global_load_async_to_lds_b128 %0, %1, off"
                 :: "v"(lds_off), "v"(g) : "memory");
}
__device__ __forceinline__ void wait_async0() {
    asm volatile("s_wait_asynccnt 0x0" ::: "memory");
}
__device__ __forceinline__ unsigned lds_addr(const void* p) {
    return (unsigned)(unsigned long long)(uintptr_t)p;  // low 32 bits = LDS offset
}

// ---------------------------------------------------------------------------
// fp32 -> bf16 conversion
// ---------------------------------------------------------------------------
__global__ void cvt_bf16(const float* __restrict__ x, __bf16* __restrict__ y, long n) {
    long i = (long)blockIdx.x * blockDim.x + threadIdx.x;
    if (i < n) y[i] = f2bf(x[i]);
}

// ---------------------------------------------------------------------------
// GEMM: C[M,N](f32) = A[M,K](bf16) * W[N,K](bf16)^T
// block = 128 threads (4 waves); block tile 128(M) x 64(N); K step 64.
// W tile (64x64) double-buffered via async-to-LDS: next K-slab's DMA is
// issued before computing the current one.  Each wave computes 32x64
// (2 A-frags x 4 B-frags -> 8 wmma per K-substep, 16 per iteration).
// ---------------------------------------------------------------------------
__global__ __launch_bounds__(128) void gemm_bf16_nt(
    const __bf16* __restrict__ A, const __bf16* __restrict__ W,
    float* __restrict__ C, int M, int N, int K) {
    __shared__ __align__(16) __bf16 sW[2][64 * 64];   // 2 x 8 KB

    const int lane = threadIdx.x & 31;
    const int wave = threadIdx.x >> 5;
    const int hl = lane >> 4;      // half of wave
    const int ln = lane & 15;
    const int kb = hl * 8;
    const int n0 = blockIdx.x * 64;
    const int m0 = blockIdx.y * 128;
    const long ar0 = (long)(m0 + wave * 32 + ln);
    const long ar1 = ar0 + 16;

    auto stage = [&](int buf, int k) {
        const unsigned swl = lds_addr(&sW[buf][0]);
#pragma unroll
        for (int i = 0; i < 4; ++i) {          // 512 16B chunks / 128 threads
            int c = i * 128 + threadIdx.x;
            int r = c >> 3, q = (c & 7) << 3;
            async_copy_b128(swl + (unsigned)c * 16,
                            &W[(long)(n0 + r) * K + k + q]);
        }
    };

    v8f acc[2][4];
#pragma unroll
    for (int u = 0; u < 2; ++u)
#pragma unroll
        for (int i = 0; i < 4; ++i) acc[u][i] = 0.f;

    stage(0, 0);
    wait_async0();
    __syncthreads();

    for (int k = 0; k < K; k += 64) {
        const int buf = (k >> 6) & 1;
        if (k + 64 < K) stage(buf ^ 1, k + 64);   // overlap DMA with wmma
#pragma unroll
        for (int ks = 0; ks < 2; ++ks) {
            v16bf a0 = load_frag(&A[ar0 * K + k + ks * 32 + kb]);
            v16bf a1 = load_frag(&A[ar1 * K + k + ks * 32 + kb]);
#pragma unroll
            for (int nt = 0; nt < 4; ++nt) {
                v16bf bfr = load_frag(&sW[buf][(nt * 16 + ln) * 64 + ks * 32 + kb]);
                acc[0][nt] = wmma_bf16(a0, bfr, acc[0][nt]);
                acc[1][nt] = wmma_bf16(a1, bfr, acc[1][nt]);
            }
        }
        wait_async0();       // next slab landed
        __syncthreads();     // everyone done reading current slab
    }
#pragma unroll
    for (int u = 0; u < 2; ++u)
#pragma unroll
        for (int nt = 0; nt < 4; ++nt)
#pragma unroll
            for (int r = 0; r < 8; ++r)
                C[(long)(m0 + wave * 32 + u * 16 + hl * 8 + r) * N +
                  n0 + nt * 16 + ln] = acc[u][nt][r];
}

// ---------------------------------------------------------------------------
// RMSNorm rows of length D: y(bf16) = w * x / sqrt(mean(x^2)+eps)
// ---------------------------------------------------------------------------
__global__ __launch_bounds__(256) void rmsnorm_bf16(
    const float* __restrict__ x, const float* __restrict__ w,
    __bf16* __restrict__ y, int D) {
    __shared__ float red[256];
    const long row = blockIdx.x;
    const float* xr = x + row * (long)D;
    float ss = 0.f;
    for (int i = threadIdx.x; i < D; i += 256) { float v = xr[i]; ss += v * v; }
    red[threadIdx.x] = ss;
    __syncthreads();
    for (int s = 128; s > 0; s >>= 1) {
        if ((int)threadIdx.x < s) red[threadIdx.x] += red[threadIdx.x + s];
        __syncthreads();
    }
    const float rinv = rsqrtf(red[0] / (float)D + 1e-6f);
    __bf16* yr = y + row * (long)D;
    for (int i = threadIdx.x; i < D; i += 256)
        yr[i] = f2bf(w[i] * xr[i] * rinv);
}

// ---------------------------------------------------------------------------
// kv prep: RMSNorm first 512 -> kvc(bf16); interleaved RoPE last 64 -> kpe(bf16)
// ---------------------------------------------------------------------------
__global__ __launch_bounds__(256) void prep_kv(
    const float* __restrict__ kv, const float* __restrict__ w,
    __bf16* __restrict__ kvc, __bf16* __restrict__ kpe) {
    __shared__ float red[256];
    const long row = blockIdx.x;
    const float* xr = kv + row * 576L;
    float ss = 0.f;
    for (int i = threadIdx.x; i < 512; i += 256) { float v = xr[i]; ss += v * v; }
    red[threadIdx.x] = ss;
    __syncthreads();
    for (int s = 128; s > 0; s >>= 1) {
        if ((int)threadIdx.x < s) red[threadIdx.x] += red[threadIdx.x + s];
        __syncthreads();
    }
    const float rinv = rsqrtf(red[0] * (1.f / 512.f) + 1e-6f);
    for (int i = threadIdx.x; i < 512; i += 256)
        kvc[row * 512 + i] = f2bf(w[i] * xr[i] * rinv);
    if (threadIdx.x < 32) {
        const int i = threadIdx.x;
        const int s = (int)(row & (S_ - 1));
        const float inv = __expf(-(float)(2 * i) * (1.f / 64.f) * 9.210340371976184f);
        float sn, cs;
        __sincosf((float)s * inv, &sn, &cs);
        const float e = xr[512 + 2 * i], od = xr[513 + 2 * i];
        kpe[row * 64 + 2 * i]     = f2bf(e * cs - od * sn);
        kpe[row * 64 + 2 * i + 1] = f2bf(e * sn + od * cs);
    }
}

// ---------------------------------------------------------------------------
// q prep: q_f32[b][s][h][192] -> qf[b][h][s][192] (bf16), RoPE on last 64,
// softmax scale folded in.  grid = (NH, B*S), block = 96.
// ---------------------------------------------------------------------------
__global__ __launch_bounds__(96) void prep_q(
    const float* __restrict__ qsrc, __bf16* __restrict__ qf) {
    const int h = blockIdx.x;
    const long bs = blockIdx.y;
    const int s = (int)(bs & (S_ - 1));
    const long b = bs >> 11;
    const float* src = qsrc + (bs * NH_ + h) * (long)DQK_;
    __bf16* dst = qf + ((b * NH_ + h) * (long)S_ + s) * DQK_;
    const int t = threadIdx.x;
    if (t < 64) {
        dst[2 * t]     = f2bf(src[2 * t]     * QK_SCALE);
        dst[2 * t + 1] = f2bf(src[2 * t + 1] * QK_SCALE);
    } else {
        const int i = t - 64;   // rope pair 0..31
        const float inv = __expf(-(float)(2 * i) * (1.f / 64.f) * 9.210340371976184f);
        float sn, cs;
        __sincosf((float)s * inv, &sn, &cs);
        const float e = src[128 + 2 * i], od = src[129 + 2 * i];
        dst[128 + 2 * i] = f2bf((e * cs - od * sn) * QK_SCALE);
        dst[129 + 2 * i] = f2bf((e * sn + od * cs) * QK_SCALE);
    }
}

// ---------------------------------------------------------------------------
// kf / v^T prep: kvx_f32[b][s][h][256](nope|v) + kpe ->
//   kf[b][h][s][192] = [k_nope | k_pe]   (bf16)
//   vt[b][h][d][s]   = v^T               (bf16, K-contiguous for PV B-frags)
// ---------------------------------------------------------------------------
__global__ __launch_bounds__(128) void prep_kf(
    const float* __restrict__ kvx, const __bf16* __restrict__ kpe,
    __bf16* __restrict__ kf, __bf16* __restrict__ vt) {
    const int h = blockIdx.x;
    const long bs = blockIdx.y;
    const int s = (int)(bs & (S_ - 1));
    const long b = bs >> 11;
    const float* src = kvx + (bs * NH_ + h) * 256L;
    __bf16* kd = kf + ((b * NH_ + h) * (long)S_ + s) * DQK_;
    const int t = threadIdx.x;
    kd[t] = f2bf(src[t]);                              // k_nope
    if (t < 64) kd[128 + t] = kpe[bs * 64 + t];        // broadcast k_pe
    vt[((b * NH_ + h) * (long)DV_ + t) * S_ + s] = f2bf(src[128 + t]);
}

// ---------------------------------------------------------------------------
// Flash attention per (b,h): 64-query blocks, 4 waves x 16 rows.
// K/V tiles double-buffered in LDS via async copies: next tile's DMA is
// issued before computing the current one (overlaps with 40 wmma / tile).
// grid = (S/64, B*NH), block = 128.
// ---------------------------------------------------------------------------
__global__ __launch_bounds__(128) void mla_attn(
    const __bf16* __restrict__ Qf, const __bf16* __restrict__ Kf,
    const __bf16* __restrict__ Vt, __bf16* __restrict__ Out) {
    __shared__ __align__(16) __bf16 sK[2][64 * DQK_];  // 2 x 24 KB
    __shared__ __align__(16) __bf16 sV[2][128 * 64];   // 2 x 16 KB
    __shared__ __align__(16) __bf16 sP[4 * 16 * 64];   // 8 KB

    const int lane = threadIdx.x & 31;
    const int wave = threadIdx.x >> 5;
    const int hl = lane >> 4, ln = lane & 15, kb = hl * 8;
    const int bh = blockIdx.y;
    const int q0 = blockIdx.x * 64;

    const __bf16* Q = Qf + (long)bh * S_ * DQK_;
    const __bf16* Kp = Kf + (long)bh * S_ * DQK_;
    const __bf16* V  = Vt + (long)bh * DV_ * S_;
    __bf16* sPw = &sP[wave * 16 * 64];

    // stage one 64-key tile (K rows are contiguous in kf -> linear copy)
    auto stage_tile = [&](int buf, int j0) {
        const unsigned kl = lds_addr(&sK[buf][0]);
        const __bf16* kg = Kp + (long)j0 * DQK_;
#pragma unroll
        for (int i = 0; i < 12; ++i) {                 // 64*192 halves
            int c = i * 128 + threadIdx.x;
            async_copy_b128(kl + (unsigned)c * 16, kg + (long)c * 8);
        }
        const unsigned vl = lds_addr(&sV[buf][0]);
#pragma unroll
        for (int i = 0; i < 8; ++i) {                  // 128*64 halves
            int c = i * 128 + threadIdx.x;
            int d = c >> 3, q = (c & 7) << 3;
            async_copy_b128(vl + (unsigned)c * 16, &V[(long)d * S_ + j0 + q]);
        }
    };

    // preload the wave's Q fragments (16 rows x 192)
    v16bf qa[6];
    const long qrow = q0 + wave * 16 + ln;
#pragma unroll
    for (int kk = 0; kk < 6; ++kk)
        qa[kk] = load_frag(&Q[qrow * DQK_ + kk * 32 + kb]);

    v8f o[8];
#pragma unroll
    for (int i = 0; i < 8; ++i) o[i] = 0.f;
    float mrow[8], lrow[8];
#pragma unroll
    for (int r = 0; r < 8; ++r) { mrow[r] = -1e30f; lrow[r] = 0.f; }

    stage_tile(0, 0);
    wait_async0();
    __syncthreads();

    for (int j0 = 0; j0 < S_; j0 += 64) {
        const int buf = (j0 >> 6) & 1;
        if (j0 + 64 < S_) stage_tile(buf ^ 1, j0 + 64);   // overlap DMA

        v8f sc[4];
#pragma unroll
        for (int i = 0; i < 4; ++i) sc[i] = 0.f;
#pragma unroll
        for (int kk = 0; kk < 6; ++kk) {
#pragma unroll
            for (int nt = 0; nt < 4; ++nt) {
                v16bf bfr = load_frag(
                    &sK[buf][(nt * 16 + ln) * DQK_ + kk * 32 + kb]);
                sc[nt] = wmma_bf16(qa[kk], bfr, sc[nt]);
            }
        }
        // online softmax: rows r+8*hl; columns spread over the 16-lane half
#pragma unroll
        for (int r = 0; r < 8; ++r) {
            float mx = mrow[r];
#pragma unroll
            for (int nt = 0; nt < 4; ++nt) mx = fmaxf(mx, sc[nt][r]);
            mx = fmaxf(mx, __shfl_xor(mx, 1));
            mx = fmaxf(mx, __shfl_xor(mx, 2));
            mx = fmaxf(mx, __shfl_xor(mx, 4));
            mx = fmaxf(mx, __shfl_xor(mx, 8));
            const float corr = __expf(mrow[r] - mx);
            float ls = 0.f;
#pragma unroll
            for (int nt = 0; nt < 4; ++nt) {
                float p = __expf(sc[nt][r] - mx);
                sc[nt][r] = p;
                ls += p;
            }
            ls += __shfl_xor(ls, 1);
            ls += __shfl_xor(ls, 2);
            ls += __shfl_xor(ls, 4);
            ls += __shfl_xor(ls, 8);
            lrow[r] = lrow[r] * corr + ls;
            mrow[r] = mx;
#pragma unroll
            for (int dt = 0; dt < 8; ++dt) o[dt][r] *= corr;
        }
        // C-layout -> A-layout via per-wave LDS tile (same-wave, in-order)
#pragma unroll
        for (int nt = 0; nt < 4; ++nt)
#pragma unroll
            for (int r = 0; r < 8; ++r)
                sPw[(hl * 8 + r) * 64 + nt * 16 + ln] = f2bf(sc[nt][r]);
        // P (16x64) * V (64x128)
#pragma unroll
        for (int kp = 0; kp < 2; ++kp) {
            v16bf pa = load_frag(&sPw[ln * 64 + kp * 32 + kb]);
#pragma unroll
            for (int dt = 0; dt < 8; ++dt) {
                v16bf vb = load_frag(
                    &sV[buf][(dt * 16 + ln) * 64 + kp * 32 + kb]);
                o[dt] = wmma_bf16(pa, vb, o[dt]);
            }
        }
        wait_async0();       // next tile landed
        __syncthreads();     // everyone done reading current tile
    }

    const int b = bh >> 4, h = bh & 15;
#pragma unroll
    for (int dt = 0; dt < 8; ++dt)
#pragma unroll
        for (int r = 0; r < 8; ++r) {
            const long row = q0 + wave * 16 + hl * 8 + r;
            Out[((long)b * S_ + row) * (NH_ * DV_) + h * DV_ + dt * 16 + ln] =
                f2bf(o[dt][r] / lrow[r]);
        }
}

// ---------------------------------------------------------------------------
// host launcher
// ---------------------------------------------------------------------------
extern "C" void kernel_launch(void* const* d_in, const int* in_sizes, int n_in,
                              void* d_out, int out_size, void* d_ws, size_t ws_size,
                              hipStream_t stream) {
    (void)in_sizes; (void)n_in; (void)out_size; (void)ws_size;
    const float* hs        = (const float*)d_in[0];
    const float* wq_a      = (const float*)d_in[1];
    const float* q_norm_w  = (const float*)d_in[2];
    const float* wq_b      = (const float*)d_in[3];
    const float* wkv_a     = (const float*)d_in[4];
    const float* kv_norm_w = (const float*)d_in[5];
    const float* wkv_b     = (const float*)d_in[6];
    const float* wo        = (const float*)d_in[7];
    float* out = (float*)d_out;

    char* base = (char*)d_ws;
    size_t off = 0;
    auto alloc = [&](size_t bytes) -> char* {
        char* p = base + off;
        off += (bytes + 255) & ~(size_t)255;
        return p;
    };
    // persistent bf16 buffers
    __bf16* hs_bf   = (__bf16*)alloc((size_t)MTOT_ * H_ * 2);
    __bf16* wqa_bf  = (__bf16*)alloc((size_t)QLORA_ * H_ * 2);
    __bf16* wqb_bf  = (__bf16*)alloc((size_t)NH_ * DQK_ * QLORA_ * 2);
    __bf16* wkva_bf = (__bf16*)alloc((size_t)(KVLORA_ + DROPE_) * H_ * 2);
    __bf16* wkvb_bf = (__bf16*)alloc((size_t)NH_ * 256 * KVLORA_ * 2);
    __bf16* wo_bf   = (__bf16*)alloc((size_t)H_ * (NH_ * DV_) * 2);
    __bf16* qf      = (__bf16*)alloc((size_t)B_ * NH_ * S_ * DQK_ * 2);
    __bf16* kf      = (__bf16*)alloc((size_t)B_ * NH_ * S_ * DQK_ * 2);
    __bf16* vt      = (__bf16*)alloc((size_t)B_ * NH_ * DV_ * S_ * 2);
    __bf16* ao_bf   = (__bf16*)alloc((size_t)MTOT_ * (NH_ * DV_) * 2);
    // phase-aliased scratch arena
    char* sc0 = base + off;
    // kv phase
    float*  kv_f32 = (float*)(sc0);
    __bf16* kvc    = (__bf16*)(sc0 + 9437184);
    __bf16* kpe    = (__bf16*)(sc0 + 9437184 + 4194304);
    float*  kvx    = (float*)(sc0 + 9437184 + 4194304 + 524288);
    // q phase (reuses same arena after kv phase is fully consumed)
    float*  qlora_f32 = (float*)(sc0);
    __bf16* qlora_bf  = (__bf16*)(sc0 + 25165824);
    float*  q_f32     = (float*)(sc0 + 25165824 + 12582912);

    auto cvt = [&](const float* s, __bf16* d, long n) {
        cvt_bf16<<<dim3((unsigned)((n + 255) / 256)), dim3(256), 0, stream>>>(s, d, n);
    };
    // 1. convert activations + weights to bf16
    cvt(hs,    hs_bf,   (long)MTOT_ * H_);
    cvt(wq_a,  wqa_bf,  (long)QLORA_ * H_);
    cvt(wq_b,  wqb_bf,  (long)NH_ * DQK_ * QLORA_);
    cvt(wkv_a, wkva_bf, (long)(KVLORA_ + DROPE_) * H_);
    cvt(wkv_b, wkvb_bf, (long)NH_ * 256 * KVLORA_);
    cvt(wo,    wo_bf,   (long)H_ * NH_ * DV_);

    // 2. kv path
    gemm_bf16_nt<<<dim3((KVLORA_ + DROPE_) / 64, MTOT_ / 128), dim3(128), 0, stream>>>(
        hs_bf, wkva_bf, kv_f32, MTOT_, KVLORA_ + DROPE_, H_);
    prep_kv<<<dim3(MTOT_), dim3(256), 0, stream>>>(kv_f32, kv_norm_w, kvc, kpe);
    gemm_bf16_nt<<<dim3((NH_ * 256) / 64, MTOT_ / 128), dim3(128), 0, stream>>>(
        kvc, wkvb_bf, kvx, MTOT_, NH_ * 256, KVLORA_);
    prep_kf<<<dim3(NH_, MTOT_), dim3(128), 0, stream>>>(kvx, kpe, kf, vt);

    // 3. q path (scratch arena reused)
    gemm_bf16_nt<<<dim3(QLORA_ / 64, MTOT_ / 128), dim3(128), 0, stream>>>(
        hs_bf, wqa_bf, qlora_f32, MTOT_, QLORA_, H_);
    rmsnorm_bf16<<<dim3(MTOT_), dim3(256), 0, stream>>>(qlora_f32, q_norm_w, qlora_bf, QLORA_);
    gemm_bf16_nt<<<dim3((NH_ * DQK_) / 64, MTOT_ / 128), dim3(128), 0, stream>>>(
        qlora_bf, wqb_bf, q_f32, MTOT_, NH_ * DQK_, QLORA_);
    prep_q<<<dim3(NH_, MTOT_), dim3(96), 0, stream>>>(q_f32, qf);

    // 4. attention
    mla_attn<<<dim3(S_ / 64, B_ * NH_), dim3(128), 0, stream>>>(qf, kf, vt, ao_bf);

    // 5. output projection -> d_out (f32)
    gemm_bf16_nt<<<dim3(H_ / 64, MTOT_ / 128), dim3(128), 0, stream>>>(
        ao_bf, wo_bf, out, MTOT_, H_, NH_ * DV_);
}
